// SimpleRNN_86998857547900
// MI455X (gfx1250) — matrix operand order repaired and checked by reading, not compiled, not run on hardware
//
#include <hip/hip_runtime.h>
#include <hip/hip_bf16.h>

// Problem constants (from reference): T=4096, D=10000, H=128, O=12, L=4
#define T_STEPS 4096
#define D_IN    10000
#define D_PAD   10016          // pad to multiple of 32 for WMMA K-steps
#define HDIM    128
#define G4      512            // 4*H gates
#define NLAYERS 4
#define O_OUT   12

typedef __attribute__((ext_vector_type(16))) __bf16 v16bf;
typedef __attribute__((ext_vector_type(8)))  __bf16 v8bf;
typedef __attribute__((ext_vector_type(8)))  float  v8f;
typedef __attribute__((ext_vector_type(4)))  float  v4f;
typedef __attribute__((ext_vector_type(4)))  unsigned int u32x4;
typedef __attribute__((ext_vector_type(4)))  int i32x4;
typedef __attribute__((ext_vector_type(8)))  int i32x8;

static __device__ __forceinline__ __bf16 f2bf(float f) { return (__bf16)f; }
static __device__ __forceinline__ float sigmoidf(float x) { return 1.0f / (1.0f + __expf(-x)); }

#if defined(__gfx1250__) && __has_builtin(__builtin_amdgcn_tensor_load_to_lds)
#define USE_TDM 1
#else
#define USE_TDM 0
#endif

#if USE_TDM
// Byte offset of a shared-memory object within the workgroup LDS allocation.
static __device__ __forceinline__ unsigned lds_byte_offset(const void* p) {
    return (unsigned)(unsigned long long)(const __attribute__((address_space(3))) void*)p;
}
#endif

// ---------------------------------------------------------------------------
// Kernel 0: fp32 weights -> bf16 copies in workspace (W_in padded to D_PAD).
// ---------------------------------------------------------------------------
__global__ void convert_weights(const float* __restrict__ W_in,
                                const float* __restrict__ W_ih,
                                const float* __restrict__ W_hh,
                                __bf16* __restrict__ Wib,
                                __bf16* __restrict__ Wih,
                                __bf16* __restrict__ Whh) {
    int stride = gridDim.x * blockDim.x;
    int idx = blockIdx.x * blockDim.x + threadIdx.x;
    const int total_wib = HDIM * D_PAD;
    for (int i = idx; i < total_wib; i += stride) {
        int r = i / D_PAD, c = i - r * D_PAD;
        Wib[i] = (c < D_IN) ? f2bf(W_in[(size_t)r * D_IN + c]) : (__bf16)0.0f;
    }
    const int total_w = NLAYERS * G4 * HDIM;
    for (int i = idx; i < total_w; i += stride) {
        Wih[i] = f2bf(W_ih[i]);
        Whh[i] = f2bf(W_hh[i]);
    }
}

// ---------------------------------------------------------------------------
// Kernel 1: X[T,H] = inputs[T,D] @ W_in.T + b_in   (bf16 WMMA, fp32 accum)
// grid = T/16 blocks, 256 threads (8 waves); wave w owns column tile w.
// ---------------------------------------------------------------------------
__global__ __launch_bounds__(256) void gemm_in(const float*  __restrict__ inputs,
                                               const __bf16* __restrict__ Wib,
                                               const float*  __restrict__ b_in,
                                               float*        __restrict__ X) {
    const int wave = threadIdx.x >> 5;
    const int lane = threadIdx.x & 31;
    const int half = lane >> 4;
    const int col  = lane & 15;
    const int m0   = blockIdx.x * 16;
    const int c0   = wave * 16;                // 8 waves * 16 = 128 columns
    const int row  = m0 + (lane & 15);

    v8f acc = {};
    const float*  arow = inputs + (size_t)row * D_IN;
    const __bf16* brow = Wib + (size_t)(c0 + col) * D_PAD;

    int kt = 0;
    for (; kt + 32 <= D_IN; kt += 32) {
        const float* ap = arow + kt + half * 8;
        v4f x0 = *(const v4f*)(ap);
        v4f x1 = *(const v4f*)(ap + 4);
        v4f x2 = *(const v4f*)(ap + 16);
        v4f x3 = *(const v4f*)(ap + 20);
        v16bf a;
        #pragma unroll
        for (int e = 0; e < 4; ++e) {
            a[e] = f2bf(x0[e]); a[4 + e] = f2bf(x1[e]);
            a[8 + e] = f2bf(x2[e]); a[12 + e] = f2bf(x3[e]);
        }
        v16bf b = __builtin_shufflevector(*(const v8bf*)(brow + kt + half * 8), *(const v8bf*)(brow + kt + half * 8 + 16), 0, 1, 2, 3, 4, 5, 6, 7, 8, 9, 10, 11, 12, 13, 14, 15);
        acc = __builtin_amdgcn_wmma_f32_16x16x32_bf16(false, a, false, b,
                                                      (short)0, acc, false, false);
    }
    {   // tail: kt = 9984; K in [9984,9999] valid for e<8 in both halves; rest zero
        const float* ap = arow + kt + half * 8;
        v4f x0 = *(const v4f*)(ap);
        v4f x1 = *(const v4f*)(ap + 4);
        v16bf a;
        #pragma unroll
        for (int e = 0; e < 4; ++e) {
            a[e] = f2bf(x0[e]); a[4 + e] = f2bf(x1[e]);
            a[8 + e] = (__bf16)0.0f; a[12 + e] = (__bf16)0.0f;
        }
        v16bf b = __builtin_shufflevector(*(const v8bf*)(brow + kt + half * 8), *(const v8bf*)(brow + kt + half * 8 + 16), 0, 1, 2, 3, 4, 5, 6, 7, 8, 9, 10, 11, 12, 13, 14, 15);   // padded region is zero
        acc = __builtin_amdgcn_wmma_f32_16x16x32_bf16(false, a, false, b,
                                                      (short)0, acc, false, false);
    }
    const float bias = b_in[c0 + col];
    #pragma unroll
    for (int r = 0; r < 8; ++r)
        X[(size_t)(m0 + half * 8 + r) * HDIM + c0 + col] = acc[r] + bias;
}

// ---------------------------------------------------------------------------
// Kernel 2 (per layer): Gin[T,4H] = seq[T,H] @ W_ih[l].T + (b_ih[l]+b_hh[l])
// grid = T/16, 256 threads (8 waves); wave w owns column tiles {w, w+8, w+16, w+24}.
// The 16x128 fp32 activation tile is staged into LDS once per block by the
// Tensor Data Mover (one tensor_load_to_lds, TDM row padding 128DW->+4DW to
// dodge LDS bank conflicts); all 8 waves then build A fragments from LDS.
// ---------------------------------------------------------------------------
#define AT_STRIDE 132   // 128 + 4 DWORD TDM pad per row

__global__ __launch_bounds__(256) void gemm_gates(const float*  __restrict__ seq,
                                                  const __bf16* __restrict__ Wih,
                                                  const float*  __restrict__ b_ih,
                                                  const float*  __restrict__ b_hh,
                                                  float*        __restrict__ Gin) {
    const int wave = threadIdx.x >> 5;
    const int lane = threadIdx.x & 31;
    const int half = lane >> 4;
    const int col  = lane & 15;
    const int m0   = blockIdx.x * 16;

#if USE_TDM
    __shared__ __attribute__((aligned(16))) float Atile[16 * AT_STRIDE];
    if (wave == 0) {
        unsigned long long ga = (unsigned long long)(const void*)(seq + (size_t)m0 * HDIM);
        u32x4 g0;
        g0[0] = 1u;                                   // count=1 (valid user descriptor)
        g0[1] = lds_byte_offset(Atile);               // lds_addr (bytes)
        g0[2] = (unsigned)(ga & 0xFFFFFFFFull);       // global_addr[31:0]
        g0[3] = (unsigned)((ga >> 32) & 0x1FFFFFFull) // global_addr[56:32]
              | (2u << 30);                           // type = 2 ("image")
        i32x8 g1;
        g1[0] = (int)((2u << 16)                      // data_size = 4 bytes
              | (1u << 20)                            // pad_enable
              | (6u << 22)                            // pad_interval: 128 DWORDs
              | (3u << 25));                          // pad_amount:   4 DWORDs
        g1[1] = (int)((128u & 0xFFFFu) << 16);        // tensor_dim0[15:0] = 128
        g1[2] = (int)((16u & 0xFFFFu) << 16);         // tensor_dim1[15:0] = 16
        g1[3] = (int)(128u << 16);                    // tile_dim0 = 128
        g1[4] = 16;                                   // tile_dim1 = 16, tile_dim2 = 0
        g1[5] = 128;                                  // tensor_dim0_stride = 128
        g1[6] = 0;                                    // tensor_dim1_stride = 0 (2D)
        g1[7] = 0;
        i32x4 z4 = (i32x4)0;
#if __clang_major__ >= 23
        i32x8 z8 = (i32x8)0;
        __builtin_amdgcn_tensor_load_to_lds(g0, g1, z4, z4, z8, 0);
#else
        __builtin_amdgcn_tensor_load_to_lds(g0, g1, z4, z4, 0);
#endif
        __builtin_amdgcn_s_wait_tensorcnt(0);
    }
    __syncthreads();
    const float* arow = Atile + (lane & 15) * AT_STRIDE;
#else
    const float* arow = seq + (size_t)(m0 + (lane & 15)) * HDIM;
#endif

    // Build the four K-fragments of A once; reuse across 4 column tiles.
    v16bf afrag[4];
    #pragma unroll
    for (int kf = 0; kf < 4; ++kf) {
        const float* ap = arow + kf * 32 + half * 8;
        v4f x0 = *(const v4f*)(ap);
        v4f x1 = *(const v4f*)(ap + 4);
        v4f x2 = *(const v4f*)(ap + 16);
        v4f x3 = *(const v4f*)(ap + 20);
        v16bf a;
        #pragma unroll
        for (int e = 0; e < 4; ++e) {
            a[e] = f2bf(x0[e]); a[4 + e] = f2bf(x1[e]);
            a[8 + e] = f2bf(x2[e]); a[12 + e] = f2bf(x3[e]);
        }
        afrag[kf] = a;
    }

    #pragma unroll
    for (int ct = 0; ct < 4; ++ct) {
        const int c0 = (ct * 8 + wave) * 16;          // 0..511
        const __bf16* brow = Wih + (size_t)(c0 + col) * HDIM;
        v8f acc = {};
        #pragma unroll
        for (int kf = 0; kf < 4; ++kf) {
            v16bf b = __builtin_shufflevector(*(const v8bf*)(brow + kf * 32 + half * 8), *(const v8bf*)(brow + kf * 32 + half * 8 + 16), 0, 1, 2, 3, 4, 5, 6, 7, 8, 9, 10, 11, 12, 13, 14, 15);
            acc = __builtin_amdgcn_wmma_f32_16x16x32_bf16(false, afrag[kf], false, b,
                                                          (short)0, acc, false, false);
        }
        const float bias = b_ih[c0 + col] + b_hh[c0 + col];
        #pragma unroll
        for (int r = 0; r < 8; ++r)
            Gin[(size_t)(m0 + half * 8 + r) * G4 + c0 + col] = acc[r] + bias;
    }
}

// ---------------------------------------------------------------------------
// Kernel 3 (per layer): sequential recurrence over T with WMMA matvec.
// Single workgroup, 512 threads = 16 waves. Each wave holds 2 row-tiles x 4
// k-fragments of W_hh resident in VGPRs. B fragment: every lane loads the same
// 16 h values -> all 16 N-columns replicate the matvec result (EXEC all-1s).
// ---------------------------------------------------------------------------
__global__ __launch_bounds__(512) void lstm_recurrence(const __bf16* __restrict__ Whh,
                                                       const float*  __restrict__ Gin,
                                                       float*        __restrict__ Yout,
                                                       float*        __restrict__ hN,
                                                       float*        __restrict__ cN) {
    __shared__ __attribute__((aligned(32))) __bf16 h_bf[HDIM];
    __shared__ float g_lds[G4];

    const int tid  = threadIdx.x;
    const int wave = tid >> 5;
    const int lane = tid & 31;
    const int half = lane >> 4;
    const int nsel = lane & 15;

    // Preload W_hh fragments (invariant across all 4096 steps).
    v16bf afrag[2][4];
    #pragma unroll
    for (int ti = 0; ti < 2; ++ti) {
        const int tile = wave * 2 + ti;
        const int arow = tile * 16 + (lane & 15);
        #pragma unroll
        for (int kf = 0; kf < 4; ++kf) {
            const __bf16* p = Whh + (size_t)arow * HDIM + kf * 32 + half * 8;
            v8bf lo = *(const v8bf*)(p);
            v8bf hi = *(const v8bf*)(p + 16);
            v16bf a;
            #pragma unroll
            for (int e = 0; e < 8; ++e) { a[e] = lo[e]; a[8 + e] = hi[e]; }
            afrag[ti][kf] = a;
        }
    }

    float c_st = 0.0f;
    if (tid < HDIM) h_bf[tid] = (__bf16)0.0f;
    __syncthreads();

    for (int t = 0; t < T_STEPS; ++t) {
        // g = W_hh @ h : 32 row-tiles x 4 K-fragments, 8 WMMAs per wave.
        v8f acc0 = {}, acc1 = {};
        #pragma unroll
        for (int kf = 0; kf < 4; ++kf) {
            v16bf b = __builtin_shufflevector(*(const v8bf*)(h_bf + kf * 32 + half * 8), *(const v8bf*)(h_bf + kf * 32 + half * 8 + 16), 0, 1, 2, 3, 4, 5, 6, 7, 8, 9, 10, 11, 12, 13, 14, 15);
            acc0 = __builtin_amdgcn_wmma_f32_16x16x32_bf16(false, afrag[0][kf], false, b,
                                                           (short)0, acc0, false, false);
            acc1 = __builtin_amdgcn_wmma_f32_16x16x32_bf16(false, afrag[1][kf], false, b,
                                                           (short)0, acc1, false, false);
        }
        // All N-columns are identical; lanes with nsel<2 scatter rows to LDS.
        if (nsel < 2) {
            const int tile = wave * 2 + nsel;
            #pragma unroll
            for (int r = 0; r < 8; ++r)
                g_lds[tile * 16 + half * 8 + r] = nsel ? acc1[r] : acc0[r];
        }
        __syncthreads();

        if (tid < HDIM) {
            const float* gp = Gin + (size_t)t * G4;      // bias already folded in
            float gi = g_lds[tid]       + gp[tid];
            float gf = g_lds[tid + 128] + gp[tid + 128];
            float gg = g_lds[tid + 256] + gp[tid + 256];
            float go = g_lds[tid + 384] + gp[tid + 384];
            gi = sigmoidf(gi); gf = sigmoidf(gf); go = sigmoidf(go);
            gg = tanhf(gg);
            c_st = gf * c_st + gi * gg;
            float hv = go * tanhf(c_st);
            h_bf[tid] = f2bf(hv);
            Yout[(size_t)t * HDIM + tid] = hv;
            if (t == T_STEPS - 1) { hN[tid] = hv; cN[tid] = c_st; }
        } else if (tid < 256 && (t + 1) < T_STEPS) {
            // pull next step's precomputed gates toward the WGP
            __builtin_prefetch(Gin + (size_t)(t + 1) * G4 + (size_t)(tid - 128) * 4, 0, 1);
        }
        __syncthreads();
    }
}

// ---------------------------------------------------------------------------
// Kernel 4: out[T,O] = sigmoid(Y[T,H] @ W_out.T + b_out)  (tiny, VALU fp32)
// ---------------------------------------------------------------------------
__global__ void out_proj(const float* __restrict__ Y, const float* __restrict__ W_out,
                         const float* __restrict__ b_out, float* __restrict__ out) {
    int idx = blockIdx.x * blockDim.x + threadIdx.x;
    if (idx >= T_STEPS * O_OUT) return;
    int t = idx / O_OUT, o = idx - t * O_OUT;
    const v4f* y = (const v4f*)(Y + (size_t)t * HDIM);
    const v4f* w = (const v4f*)(W_out + (size_t)o * HDIM);
    float acc = 0.0f;
    #pragma unroll
    for (int k = 0; k < HDIM / 4; ++k) {
        v4f a = y[k], b = w[k];
        acc += a[0] * b[0] + a[1] * b[1] + a[2] * b[2] + a[3] * b[3];
    }
    out[idx] = sigmoidf(acc + b_out[o]);
}

// ---------------------------------------------------------------------------
extern "C" void kernel_launch(void* const* d_in, const int* in_sizes, int n_in,
                              void* d_out, int out_size, void* d_ws, size_t ws_size,
                              hipStream_t stream) {
    const float* inputs = (const float*)d_in[0];
    const float* W_in   = (const float*)d_in[1];
    const float* b_in   = (const float*)d_in[2];
    const float* W_ih   = (const float*)d_in[3];
    const float* W_hh   = (const float*)d_in[4];
    const float* b_ih   = (const float*)d_in[5];
    const float* b_hh   = (const float*)d_in[6];
    const float* W_out  = (const float*)d_in[7];
    const float* b_out  = (const float*)d_in[8];
    float* out = (float*)d_out;

    char* ws = (char*)d_ws;
    size_t off = 0;
    __bf16* Wib = (__bf16*)(ws + off); off += (size_t)HDIM * D_PAD * 2;        // 2.56 MB
    __bf16* Wih = (__bf16*)(ws + off); off += (size_t)NLAYERS * G4 * HDIM * 2; // 0.5 MB
    __bf16* Whh = (__bf16*)(ws + off); off += (size_t)NLAYERS * G4 * HDIM * 2; // 0.5 MB
    float*  buf0 = (float*)(ws + off); off += (size_t)T_STEPS * HDIM * 4;      // 2 MB
    float*  buf1 = (float*)(ws + off); off += (size_t)T_STEPS * HDIM * 4;      // 2 MB
    float*  Gin  = (float*)(ws + off); off += (size_t)T_STEPS * G4 * 4;        // 8 MB

    convert_weights<<<512, 256, 0, stream>>>(W_in, W_ih, W_hh, Wib, Wih, Whh);
    gemm_in<<<T_STEPS / 16, 256, 0, stream>>>(inputs, Wib, b_in, buf0);

    float* hOut = out + (size_t)T_STEPS * O_OUT;
    float* cOut = hOut + NLAYERS * HDIM;
    float* seqIn = buf0;
    float* seqOut = buf1;
    for (int l = 0; l < NLAYERS; ++l) {
        gemm_gates<<<T_STEPS / 16, 256, 0, stream>>>(
            seqIn, Wih + (size_t)l * G4 * HDIM, b_ih + l * G4, b_hh + l * G4, Gin);
        lstm_recurrence<<<1, 512, 0, stream>>>(
            Whh + (size_t)l * G4 * HDIM, Gin, seqOut, hOut + l * HDIM, cOut + l * HDIM);
        float* tmp = seqIn; seqIn = seqOut; seqOut = tmp;
    }
    out_proj<<<(T_STEPS * O_OUT + 255) / 256, 256, 0, stream>>>(seqIn, W_out, b_out, out);
}
